// Seq2SeqWithAttention_77034533421109
// MI455X (gfx1250) — matrix operand order repaired and checked by reading, not compiled, not run on hardware
//
#include <hip/hip_runtime.h>
#include <hip/hip_bf16.h>

// Model dims
#define B_   128
#define TE_  256
#define TD_  32
#define FE_  8
#define FD_  7
#define UE_  256
#define UD_  256
#define DC_  128

typedef __attribute__((ext_vector_type(16))) __bf16 v16bf;
typedef __attribute__((ext_vector_type(8)))  float  v8f;

__device__ __forceinline__ v8f wmma_bf16(v16bf a, v16bf b, v8f c) {
  // D = A(16x32 bf16) * B(32x16 bf16) + C(16x16 f32)
  return __builtin_amdgcn_wmma_f32_16x16x32_bf16(false, a, false, b, (short)0, c, false, false);
}

// A fragment: 16x32 bf16 tile (rows m0..m0+15, cols kbase..kbase+31) of row-major A (lda elems).
// ISA 16-bit A layout: lane%16 = M; VGPR r: K = 16*(r/4) + 8*(lane/16) + 2*(r%4) (+kbase).
__device__ __forceinline__ v16bf load_a_frag(const __bf16* A, int lda, int m0, int kbase, int lane) {
  const __bf16* row = A + (size_t)(m0 + (lane & 15)) * lda;
  const int half8 = (lane >> 4) << 3;
  v16bf a;
#pragma unroll
  for (int r = 0; r < 8; ++r) {
    int k = kbase + ((r >> 2) << 4) + half8 + ((r & 3) << 1);
    a[2 * r]     = row[k];
    a[2 * r + 1] = row[k + 1];
  }
  return a;
}

// B fragment: 32x16 bf16 tile (rows kbase.., cols n0..) of row-major B (ldb elems).
// lane%16 = N; VGPR r: K = 16*(r/4) + 8*(lane/16) + 2*(r%4) (+kbase), same as A.
__device__ __forceinline__ v16bf load_b_frag(const __bf16* B, int ldb, int kbase, int n0, int lane) {
  const __bf16* col = B + n0 + (lane & 15);
  const int half8 = (lane >> 4) << 3;
  v16bf b;
#pragma unroll
  for (int r = 0; r < 8; ++r) {
    int k = kbase + ((r >> 2) << 4) + half8 + ((r & 3) << 1);
    b[2 * r]     = col[(size_t)k * ldb];
    b[2 * r + 1] = col[(size_t)(k + 1) * ldb];
  }
  return b;
}

__device__ __forceinline__ float sigf(float x) { return 1.f / (1.f + __expf(-x)); }
// Branchless tanh: exact saturation at both ends (t->inf => 1, t->0 => -1), no EXEC divergence.
__device__ __forceinline__ float fast_tanh(float x) {
  float t = __expf(2.f * x);
  return 1.f - 2.f / (t + 1.f);
}

// ---------------- elementwise f32 -> bf16 convert ----------------
__global__ void cvt_bf16_kernel(const float* __restrict__ src, __bf16* __restrict__ dst, int n) {
  int i = blockIdx.x * blockDim.x + threadIdx.x;
  if (i < n) dst[i] = (__bf16)src[i];
}

// ---------------- pointwise conv compression (FE/FD -> 64 -> DC, ReLU) ----------------
__global__ void compress_kernel(const float* __restrict__ x, int fin,
                                const float* __restrict__ w1, const float* __restrict__ b1,
                                const float* __restrict__ w2, const float* __restrict__ b2,
                                __bf16* __restrict__ out, int rows) {
  int row = blockIdx.x * blockDim.x + threadIdx.x;
  if (row >= rows) return;
  const float* xr = x + (size_t)row * fin;
  float hid[64];
#pragma unroll 4
  for (int h = 0; h < 64; ++h) {
    float acc = b1[h];
    for (int f = 0; f < fin; ++f) acc += xr[f] * w1[f * 64 + h];
    hid[h] = fmaxf(acc, 0.f);
  }
  __bf16* orow = out + (size_t)row * DC_;
  for (int o = 0; o < DC_; ++o) {
    float acc = b2[o];
#pragma unroll 8
    for (int h = 0; h < 64; ++h) acc += hid[h] * w2[h * DC_ + o];
    orow[o] = (__bf16)fmaxf(acc, 0.f);
  }
}

// ---------------- persistent bidirectional encoder LSTM ----------------
// grid.x = 2 (dir), 1024 threads = 32 waves, one WGP per direction.
// h resident in LDS (bf16, 64KB of 320KB); c resident in VGPRs across all 256 steps.
// Each wave owns 4 gate-quadruples: bt = wave&7, u-tiles (wave>>3)*4 .. +3.
__global__ void __launch_bounds__(1024)
lstm_dir_kernel(const __bf16* __restrict__ ec,
                const __bf16* __restrict__ f_wx, const __bf16* __restrict__ f_wh, const float* __restrict__ f_b,
                const __bf16* __restrict__ b_wx, const __bf16* __restrict__ b_wh, const float* __restrict__ b_b,
                __bf16* __restrict__ enc_out,   // (B,TE,2UE) bf16
                __bf16* __restrict__ hcat,      // (B,2UE) bf16 final h
                __bf16* __restrict__ ccat)      // (B,2UE) bf16 final c
{
  __shared__ __bf16 h_lds[B_ * UE_];  // 64 KB
  const int dir = blockIdx.x;
  const __bf16* wx = dir ? b_wx : f_wx;
  const __bf16* wh = dir ? b_wh : f_wh;
  const float*  bb = dir ? b_b : f_b;

  const int tid  = threadIdx.x;
  const int lane = tid & 31;
  const int wave = tid >> 5;
  const int bt   = wave & 7;
  const int ut0  = (wave >> 3) << 2;

  for (int i = tid; i < B_ * UE_; i += 1024) h_lds[i] = (__bf16)0.f;
  __syncthreads();

  const int colq = lane & 15;
  const int rowq = (lane >> 4) << 3;

  // hoist per-gate biases out of the 256-step time loop
  float bgate[4][4];
#pragma unroll
  for (int q = 0; q < 4; ++q)
#pragma unroll
    for (int g = 0; g < 4; ++g) bgate[q][g] = bb[g * UE_ + (ut0 + q) * 16 + colq];

  float creg[4][8];
  float hreg[4][8];
#pragma unroll
  for (int q = 0; q < 4; ++q)
#pragma unroll
    for (int r = 0; r < 8; ++r) { creg[q][r] = 0.f; hreg[q][r] = 0.f; }

  for (int t = 0; t < TE_; ++t) {
    const int tt = dir ? (TE_ - 1 - t) : t;
    const __bf16* xt = ec + (size_t)tt * DC_;  // row stride TE_*DC_
#pragma unroll 1
    for (int q = 0; q < 4; ++q) {
      const int ut   = ut0 + q;
      const int ncol = ut * 16;
      v8f acc[4];
#pragma unroll
      for (int g = 0; g < 4; ++g) {
#pragma unroll
        for (int r = 0; r < 8; ++r) acc[g][r] = bgate[q][g];
      }
      // x_t @ wx (K = DC)
#pragma unroll 1
      for (int kb = 0; kb < DC_; kb += 32) {
        v16bf a = load_a_frag(xt, TE_ * DC_, bt * 16, kb, lane);
        __builtin_prefetch(wx + (size_t)(kb + 32) * (4 * UE_), 0, 1);
#pragma unroll
        for (int g = 0; g < 4; ++g) {
          v16bf bf = load_b_frag(wx, 4 * UE_, kb, g * UE_ + ncol, lane);
          acc[g] = wmma_bf16(a, bf, acc[g]);
        }
      }
      // h_{t-1} @ wh (K = UE), A streamed from LDS
#pragma unroll 1
      for (int kb = 0; kb < UE_; kb += 32) {
        v16bf a = load_a_frag(h_lds, UE_, bt * 16, kb, lane);
        __builtin_prefetch(wh + (size_t)(kb + 32) * (4 * UE_), 0, 1);
#pragma unroll
        for (int g = 0; g < 4; ++g) {
          v16bf bf = load_b_frag(wh, 4 * UE_, kb, g * UE_ + ncol, lane);
          acc[g] = wmma_bf16(a, bf, acc[g]);
        }
      }
      // Keras gate order i, f, g, o — branchless, fully in registers on the acc layout.
#pragma unroll
      for (int r = 0; r < 8; ++r) {
        float iv = sigf(acc[0][r]);
        float fv = sigf(acc[1][r]);
        float gv = fast_tanh(acc[2][r]);
        float ov = sigf(acc[3][r]);
        float cn = fv * creg[q][r] + iv * gv;
        creg[q][r] = cn;
        hreg[q][r] = ov * fast_tanh(cn);
      }
    }
    __syncthreads();  // all waves finished reading old h
#pragma unroll
    for (int q = 0; q < 4; ++q) {
      const int ut = ut0 + q;
#pragma unroll
      for (int r = 0; r < 8; ++r) {
        int b = bt * 16 + rowq + r;
        int u = ut * 16 + colq;
        __bf16 hv = (__bf16)hreg[q][r];
        h_lds[b * UE_ + u] = hv;
        enc_out[((size_t)b * TE_ + tt) * (2 * UE_) + dir * UE_ + u] = hv;
      }
    }
    __syncthreads();
  }
#pragma unroll
  for (int q = 0; q < 4; ++q) {
    const int ut = ut0 + q;
#pragma unroll
    for (int r = 0; r < 8; ++r) {
      int b = bt * 16 + rowq + r;
      int u = ut * 16 + colq;
      hcat[b * (2 * UE_) + dir * UE_ + u] = (__bf16)hreg[q][r];
      ccat[b * (2 * UE_) + dir * UE_ + u] = (__bf16)creg[q][r];
    }
  }
}

// ---------------- generic bf16 WMMA GEMM: C = A*B + bias (f32 out, optional bf16 mirror) ----------------
__global__ void gemm_bf16_kernel(const __bf16* __restrict__ A, int lda,
                                 const __bf16* __restrict__ Bm, int ldb,
                                 const float* __restrict__ bias,
                                 float* __restrict__ C, int ldc,
                                 __bf16* __restrict__ Cbf, int ldcb,
                                 int M, int N, int K) {
  int gwave = (blockIdx.x * blockDim.x + threadIdx.x) >> 5;
  int lane  = threadIdx.x & 31;
  int ntn   = N >> 4;
  int mt    = gwave / ntn;
  int nt    = gwave - mt * ntn;
  if (mt * 16 >= M) return;  // wave-uniform: EXEC stays all-ones for WMMA
  const int colq = lane & 15;
  const int rowq = (lane >> 4) << 3;
  v8f acc;
  float bv = bias ? bias[nt * 16 + colq] : 0.f;
#pragma unroll
  for (int r = 0; r < 8; ++r) acc[r] = bv;
  for (int kb = 0; kb < K; kb += 32) {
    v16bf a = load_a_frag(A, lda, mt * 16, kb, lane);
    v16bf b = load_b_frag(Bm, ldb, kb, nt * 16, lane);
    __builtin_prefetch(Bm + (size_t)(kb + 32) * ldb + nt * 16, 0, 1);
    acc = wmma_bf16(a, b, acc);
  }
#pragma unroll
  for (int r = 0; r < 8; ++r) {
    int m = mt * 16 + rowq + r;
    int n = nt * 16 + colq;
    C[(size_t)m * ldc + n] = acc[r];
    if (Cbf) Cbf[(size_t)m * ldcb + n] = (__bf16)acc[r];
  }
}

// ---------------- fused attention step: hq -> scores -> softmax -> ctx -> xcat ----------------
// One block (256 threads = 8 waves) per batch row. Everything staged through LDS;
// no hq/score global round-trips, 1 launch instead of 3 per decoder step.
__global__ void __launch_bounds__(256)
att_step_kernel(const float* __restrict__ h_f32,    // (B,UD) current decoder h
                const float* __restrict__ att_w2,   // (UD,UD)
                const float* __restrict__ att_b2,   // (UD)
                const float* __restrict__ att_enc,  // (B,TE,UD)
                const float* __restrict__ att_v,    // (UD)
                const float* __restrict__ att_vb,   // (1)
                const __bf16* __restrict__ enc_out, // (B,TE,2UE)
                const __bf16* __restrict__ dcmp,    // (B,TD,DC)
                int t, __bf16* __restrict__ xcat)   // (B,640)
{
  __shared__ float hs[UD_];
  __shared__ float hqs[UD_];
  __shared__ float w[TE_];
  __shared__ float red[TE_];
  const int b = blockIdx.x, tid = threadIdx.x;
  const int lane = tid & 31, wv = tid >> 5;

  hs[tid] = h_f32[b * UD_ + tid];
  __syncthreads();

  // hq[u] = h . att_w2[:,u] + att_b2[u]  (column access coalesced across tid)
  {
    float acc = att_b2[tid];
    for (int k = 0; k < UD_; ++k) acc += hs[k] * att_w2[k * UD_ + tid];
    hqs[tid] = acc;
  }
  __syncthreads();

  // scores: wave per te (lane-strided u reads of att_enc are coalesced)
  const float vb = att_vb[0];
  for (int te = wv; te < TE_; te += 8) {
    const float* ae = att_enc + ((size_t)b * TE_ + te) * UD_;
    float s = 0.f;
    for (int u = lane; u < UD_; u += 32) s += fast_tanh(ae[u] + hqs[u]) * att_v[u];
#pragma unroll
    for (int off = 16; off; off >>= 1) s += __shfl_xor(s, off, 32);
    if (!lane) w[te] = s + vb;
  }
  __syncthreads();

  // softmax over TE
  float s = w[tid];
  red[tid] = s;
  __syncthreads();
  for (int off = 128; off > 0; off >>= 1) {
    if (tid < off) red[tid] = fmaxf(red[tid], red[tid + off]);
    __syncthreads();
  }
  float m = red[0];
  __syncthreads();
  float e = __expf(s - m);
  w[tid] = e;
  red[tid] = e;
  __syncthreads();
  for (int off = 128; off > 0; off >>= 1) {
    if (tid < off) red[tid] += red[tid + off];
    __syncthreads();
  }
  float inv = 1.f / red[0];

  // context + xcat = [ctx(512) bf16, x_t(128) bf16]
  float acc0 = 0.f, acc1 = 0.f;
  for (int te = 0; te < TE_; ++te) {
    float wt = w[te];
    const __bf16* er = enc_out + ((size_t)b * TE_ + te) * (2 * UE_);
    acc0 += wt * (float)er[tid];
    acc1 += wt * (float)er[UE_ + tid];
  }
  xcat[b * 640 + tid]       = (__bf16)(acc0 * inv);
  xcat[b * 640 + UE_ + tid] = (__bf16)(acc1 * inv);
  if (tid < DC_) xcat[b * 640 + 2 * UE_ + tid] = dcmp[((size_t)b * TD_ + t) * DC_ + tid];
}

// ---------------- decoder LSTM cell: z = xcat@dl_wx + h@dl_wh + b; gates; h,c update ----------------
__global__ void dec_lstm_kernel(const __bf16* __restrict__ xcat,  // (B,640)
                                const __bf16* __restrict__ wx,    // (640,1024)
                                const __bf16* __restrict__ hin,   // (B,256)
                                const __bf16* __restrict__ wh,    // (256,1024)
                                const float* __restrict__ bias,   // (1024)
                                float* __restrict__ h_f32,
                                __bf16* __restrict__ hout,
                                float* __restrict__ c_f32) {
  int gwave = (blockIdx.x * blockDim.x + threadIdx.x) >> 5;
  int lane  = threadIdx.x & 31;
  int bt = gwave & 7;
  int ut = gwave >> 3;
  const int colq = lane & 15, rowq = (lane >> 4) << 3;
  const int ncol = ut * 16;
  v8f acc[4];
#pragma unroll
  for (int g = 0; g < 4; ++g) {
    float bv = bias[g * UD_ + ncol + colq];
#pragma unroll
    for (int r = 0; r < 8; ++r) acc[g][r] = bv;
  }
#pragma unroll 1
  for (int kb = 0; kb < 640; kb += 32) {
    v16bf a = load_a_frag(xcat, 640, bt * 16, kb, lane);
    __builtin_prefetch(wx + (size_t)(kb + 32) * (4 * UD_), 0, 1);
#pragma unroll
    for (int g = 0; g < 4; ++g) {
      v16bf bf = load_b_frag(wx, 4 * UD_, kb, g * UD_ + ncol, lane);
      acc[g] = wmma_bf16(a, bf, acc[g]);
    }
  }
#pragma unroll 1
  for (int kb = 0; kb < UD_; kb += 32) {
    v16bf a = load_a_frag(hin, UD_, bt * 16, kb, lane);
    __builtin_prefetch(wh + (size_t)(kb + 32) * (4 * UD_), 0, 1);
#pragma unroll
    for (int g = 0; g < 4; ++g) {
      v16bf bf = load_b_frag(wh, 4 * UD_, kb, g * UD_ + ncol, lane);
      acc[g] = wmma_bf16(a, bf, acc[g]);
    }
  }
#pragma unroll
  for (int r = 0; r < 8; ++r) {
    int b = bt * 16 + rowq + r;
    int u = ncol + colq;
    float cold = c_f32[b * UD_ + u];
    float iv = sigf(acc[0][r]);
    float fv = sigf(acc[1][r]);
    float gv = fast_tanh(acc[2][r]);
    float ov = sigf(acc[3][r]);
    float cn = fv * cold + iv * gv;
    float hn = ov * fast_tanh(cn);
    c_f32[b * UD_ + u] = cn;
    h_f32[b * UD_ + u] = hn;
    hout[b * UD_ + u]  = (__bf16)hn;
  }
}

// ---------------- pred = h @ fc_w + fc_b ----------------
__global__ void pred_kernel(const float* __restrict__ h, const float* __restrict__ fc_w,
                            const float* __restrict__ fc_b, float* __restrict__ out, int t) {
  int gwave = (blockIdx.x * blockDim.x + threadIdx.x) >> 5;
  int lane  = threadIdx.x & 31;
  if (gwave >= B_) return;
  float s = 0.f;
  for (int u = lane; u < UD_; u += 32) s += h[gwave * UD_ + u] * fc_w[u];
#pragma unroll
  for (int off = 16; off; off >>= 1) s += __shfl_xor(s, off, 32);
  if (!lane) out[gwave * TD_ + t] = s + fc_b[0];
}

extern "C" void kernel_launch(void* const* d_in, const int* in_sizes, int n_in,
                              void* d_out, int out_size, void* d_ws, size_t ws_size,
                              hipStream_t stream) {
  (void)in_sizes; (void)n_in; (void)out_size; (void)ws_size;
  const float* enc_x  = (const float*)d_in[0];
  const float* dec_x  = (const float*)d_in[1];
  const float* ec1_w  = (const float*)d_in[2];
  const float* ec1_b  = (const float*)d_in[3];
  const float* ec2_w  = (const float*)d_in[4];
  const float* ec2_b  = (const float*)d_in[5];
  const float* dc1_w  = (const float*)d_in[6];
  const float* dc1_b  = (const float*)d_in[7];
  const float* dc2_w  = (const float*)d_in[8];
  const float* dc2_b  = (const float*)d_in[9];
  const float* lf_wx  = (const float*)d_in[10];
  const float* lf_wh  = (const float*)d_in[11];
  const float* lf_b   = (const float*)d_in[12];
  const float* lb_wx  = (const float*)d_in[13];
  const float* lb_wh  = (const float*)d_in[14];
  const float* lb_b   = (const float*)d_in[15];
  const float* st_w   = (const float*)d_in[16];
  const float* st_b   = (const float*)d_in[17];
  const float* att_w1 = (const float*)d_in[18];
  const float* att_b1 = (const float*)d_in[19];
  const float* att_w2 = (const float*)d_in[20];
  const float* att_b2 = (const float*)d_in[21];
  const float* att_v  = (const float*)d_in[22];
  const float* att_vb = (const float*)d_in[23];
  const float* dl_wx  = (const float*)d_in[24];
  const float* dl_wh  = (const float*)d_in[25];
  const float* dl_b   = (const float*)d_in[26];
  const float* fc_w   = (const float*)d_in[27];
  const float* fc_b   = (const float*)d_in[28];
  float* out = (float*)d_out;

  // -------- workspace carve-out (all buffers written before read) --------
  char* ws = (char*)d_ws;
  size_t off = 0;
  auto alloc = [&](size_t bytes) -> void* {
    off = (off + 255) & ~(size_t)255;
    void* p = ws + off;
    off += bytes;
    return p;
  };
  __bf16* ec_bf   = (__bf16*)alloc((size_t)B_ * TE_ * DC_ * 2);
  __bf16* dc_bf   = (__bf16*)alloc((size_t)B_ * TD_ * DC_ * 2);
  __bf16* lfwx_bf = (__bf16*)alloc((size_t)DC_ * 4 * UE_ * 2);
  __bf16* lfwh_bf = (__bf16*)alloc((size_t)UE_ * 4 * UE_ * 2);
  __bf16* lbwx_bf = (__bf16*)alloc((size_t)DC_ * 4 * UE_ * 2);
  __bf16* lbwh_bf = (__bf16*)alloc((size_t)UE_ * 4 * UE_ * 2);
  __bf16* stw_bf  = (__bf16*)alloc((size_t)2 * UE_ * UD_ * 2);
  __bf16* att1_bf = (__bf16*)alloc((size_t)2 * UE_ * UD_ * 2);
  __bf16* dlwx_bf = (__bf16*)alloc((size_t)640 * 4 * UD_ * 2);
  __bf16* dlwh_bf = (__bf16*)alloc((size_t)UD_ * 4 * UD_ * 2);
  __bf16* enc_out = (__bf16*)alloc((size_t)B_ * TE_ * 2 * UE_ * 2);
  float*  att_enc = (float*) alloc((size_t)B_ * TE_ * UD_ * 4);
  __bf16* hcat    = (__bf16*)alloc((size_t)B_ * 2 * UE_ * 2);
  __bf16* ccat    = (__bf16*)alloc((size_t)B_ * 2 * UE_ * 2);
  float*  h_f32   = (float*) alloc((size_t)B_ * UD_ * 4);
  float*  c_f32   = (float*) alloc((size_t)B_ * UD_ * 4);
  __bf16* hbuf0   = (__bf16*)alloc((size_t)B_ * UD_ * 2);
  __bf16* hbuf1   = (__bf16*)alloc((size_t)B_ * UD_ * 2);
  __bf16* xcat    = (__bf16*)alloc((size_t)B_ * 640 * 2);

  auto cvt = [&](const float* s, __bf16* d, int n) {
    cvt_bf16_kernel<<<(n + 255) / 256, 256, 0, stream>>>(s, d, n);
  };
  auto gemm = [&](const __bf16* A, int lda, const __bf16* Bm, int ldb, const float* bias,
                  float* C, int ldc, __bf16* Cbf, int ldcb, int M, int N, int K) {
    int waves = (M / 16) * (N / 16);
    gemm_bf16_kernel<<<(waves * 32 + 255) / 256, 256, 0, stream>>>(A, lda, Bm, ldb, bias, C,
                                                                   ldc, Cbf, ldcb, M, N, K);
  };

  // weight conversions (f32 -> bf16); weights stay L2-resident for the whole run
  cvt(lf_wx, lfwx_bf, DC_ * 4 * UE_);
  cvt(lf_wh, lfwh_bf, UE_ * 4 * UE_);
  cvt(lb_wx, lbwx_bf, DC_ * 4 * UE_);
  cvt(lb_wh, lbwh_bf, UE_ * 4 * UE_);
  cvt(st_w,  stw_bf,  2 * UE_ * UD_);
  cvt(att_w1, att1_bf, 2 * UE_ * UD_);
  cvt(dl_wx, dlwx_bf, 640 * 4 * UD_);
  cvt(dl_wh, dlwh_bf, UD_ * 4 * UD_);

  // feature compression
  compress_kernel<<<(B_ * TE_) / 256, 256, 0, stream>>>(enc_x, FE_, ec1_w, ec1_b, ec2_w, ec2_b,
                                                        ec_bf, B_ * TE_);
  compress_kernel<<<(B_ * TD_) / 256, 256, 0, stream>>>(dec_x, FD_, dc1_w, dc1_b, dc2_w, dc2_b,
                                                        dc_bf, B_ * TD_);

  // persistent bidirectional encoder LSTM (one WGP per direction)
  lstm_dir_kernel<<<2, 1024, 0, stream>>>(ec_bf, lfwx_bf, lfwh_bf, lf_b, lbwx_bf, lbwh_bf, lb_b,
                                          enc_out, hcat, ccat);

  // att_enc = enc_out @ att_w1 + att_b1   (32768 x 512 x 256)
  gemm(enc_out, 2 * UE_, att1_bf, UD_, att_b1, att_enc, UD_, nullptr, 0, B_ * TE_, UD_, 2 * UE_);

  // decoder initial state: h = hcat@st_w + st_b ; c = ccat@st_w + st_b
  gemm(hcat, 2 * UE_, stw_bf, UD_, st_b, h_f32, UD_, hbuf0, UD_, B_, UD_, 2 * UE_);
  gemm(ccat, 2 * UE_, stw_bf, UD_, st_b, c_f32, UD_, nullptr, 0, B_, UD_, 2 * UE_);

  // decoder time loop: 3 launches per step (fused attention, LSTM cell, pred)
  for (int t = 0; t < TD_; ++t) {
    __bf16* hin  = (t & 1) ? hbuf1 : hbuf0;
    __bf16* hout = (t & 1) ? hbuf0 : hbuf1;
    att_step_kernel<<<B_, 256, 0, stream>>>(h_f32, att_w2, att_b2, att_enc, att_v, att_vb,
                                            enc_out, dc_bf, t, xcat);
    dec_lstm_kernel<<<16, 256, 0, stream>>>(xcat, dlwx_bf, hin, dlwh_bf, dl_b, h_f32, hout, c_f32);
    pred_kernel<<<16, 256, 0, stream>>>(h_f32, fc_w, fc_b, out, t);
  }
}
